// Actor_47588237640247
// MI455X (gfx1250) — compile-verified, matrix operand order checked by reading, not run on hardware
//
#include <hip/hip_runtime.h>
#include <hip/hip_bf16.h>
#include <math.h>

typedef __attribute__((ext_vector_type(16))) _Float16 v16h;
typedef __attribute__((ext_vector_type(8)))  _Float16 v8h;
typedef __attribute__((ext_vector_type(8)))  float    v8f;
typedef __attribute__((ext_vector_type(8)))  float    f32x8;

#define BB  64
#define LL  512
#define EE  256
#define HH  128
#define CC  128
#define TT  511
#define FF  768
#define LP2 514

// Workspace layout (bytes)
#define WS_CF    0                         // 512*64*768 f16 = 50331648
#define WS_L1W   50331648                  // 768*256 f16
#define WS_P1W   50724864                  // 128*256 f16
#define WS_WIH   50790400                  // 512*256 f16
#define WS_WHH   51052544                  // 512*128 f16

// Dynamic LDS layout (bytes) for the scan kernel (total 300032 <= 320K WGP cap)
#define LDS_H16    0                       // 64*128 f16   = 16384
#define LDS_C      16384                   // 64*128 f32   = 32768
#define LDS_SC     49152                   // 384 f32      = 1536
#define LDS_ACC    50688                   // 64 f32       = 256
#define LDS_IDX    50944                   // 64 int       = 256
#define LDS_P2W    51200                   // 128 f32      = 512
#define LDS_P1B    51712                   // 128 f32      = 512
#define LDS_BSUM   52224                   // 512 f32      = 2048
#define LDS_CHOSEN 54272                   // 64*384 f16   = 49152
#define LDS_P1W    103424                  // 128*256 f16  = 65536
#define LDS_UNION  168960                  // max(hid 98304, gates 131072) = 131072
#define LDS_BYTES  300032

// ---- Fragment loaders -------------------------------------------------------
// 16-bit A/B fragment for v_wmma_f32_16x16x32_f16: per lane, two contiguous
// 16-byte runs at K = khalf*8 .. +7 and K = 16+khalf*8 .. +7.
__device__ __forceinline__ v16h frag_ld16(const _Float16* base, int khalf) {
    v8h lo = *(const v8h*)(base + khalf * 8);
    v8h hi = *(const v8h*)(base + 16 + khalf * 8);
    return __builtin_shufflevector(lo, hi, 0, 1, 2, 3, 4, 5, 6, 7,
                                   8, 9, 10, 11, 12, 13, 14, 15);
}

__device__ __forceinline__ v16h frag_ld_f32(const float* base, int khalf) {
    f32x8 lo = *(const f32x8*)(base + khalf * 8);
    f32x8 hi = *(const f32x8*)(base + 16 + khalf * 8);
    v16h r;
#pragma unroll
    for (int j = 0; j < 8; ++j) {
        r[j]     = (_Float16)lo[j];
        r[8 + j] = (_Float16)hi[j];
    }
    return r;
}

#define WMMA(acc, a, b) \
    __builtin_amdgcn_wmma_f32_16x16x32_f16(false, (a), false, (b), (short)0, (acc), false, false)

// ---- f32 -> f16 conversion --------------------------------------------------
__global__ void cvt_f16_kernel(const float* __restrict__ src,
                               _Float16* __restrict__ dst, int n) {
    int i = blockIdx.x * blockDim.x + threadIdx.x;
    if (i < n) dst[i] = (_Float16)src[i];
}

// ---- Phase 1: cf = relu(memory[:,1:-1] @ l1_w^T + l1_b) ---------------------
__global__ __launch_bounds__(256)
void cf_gemm_kernel(const float* __restrict__ mem,
                    const _Float16* __restrict__ l1w16,
                    const float* __restrict__ l1b,
                    _Float16* __restrict__ cf16) {
    int wave = threadIdx.x >> 5;
    int lane = threadIdx.x & 31;
    int tile = blockIdx.x * 8 + wave;
    int tm = tile / 48, tn = tile % 48;
    int r0 = tm * 16, c0 = tn * 16;
    int m16   = lane & 15;
    int khalf = lane >> 4;

    int g   = r0 + m16;
    int b   = g >> 9;
    int pos = g & 511;
    const float*    arow = mem   + ((size_t)b * LP2 + pos + 1) * EE;
    const _Float16* brow = l1w16 + (size_t)(c0 + m16) * EE;

    v8f acc = {};
#pragma unroll
    for (int kb = 0; kb < EE; kb += 32) {
        v16h a  = frag_ld_f32(arow + kb, khalf);
        v16h bf = frag_ld16(brow + kb, khalf);
        acc = WMMA(acc, a, bf);
    }
#pragma unroll
    for (int v = 0; v < 8; ++v) {
        int mm = v + khalf * 8;
        int gg = r0 + mm;
        int bb = gg >> 9, pp = gg & 511;
        float val = acc[v] + l1b[c0 + m16];
        val = val > 0.f ? val : 0.f;
        cf16[((size_t)pp * BB + bb) * FF + c0 + m16] = (_Float16)val;
    }
}

// ---- Phase 2: persistent sequential scan ------------------------------------
__global__ __launch_bounds__(512)
void scan_kernel(const _Float16* __restrict__ cf16,
                 const _Float16* __restrict__ p1w16,
                 const _Float16* __restrict__ wih16,
                 const _Float16* __restrict__ whh16,
                 const float* __restrict__ p1b,
                 const float* __restrict__ p2w,
                 const float* __restrict__ p2b,
                 const float* __restrict__ bih,
                 const float* __restrict__ bhh,
                 const int* __restrict__ maskp,
                 const int* __restrict__ lengthp,
                 const float* __restrict__ gumbel,
                 float* __restrict__ out) {
    extern __shared__ char smem[];
    _Float16* h16    = (_Float16*)(smem + LDS_H16);
    float*    cmem   = (float*)   (smem + LDS_C);
    float*    scores = (float*)   (smem + LDS_SC);
    float*    accv   = (float*)   (smem + LDS_ACC);
    int*      cidx   = (int*)     (smem + LDS_IDX);
    float*    p2wl   = (float*)   (smem + LDS_P2W);
    float*    p1bl   = (float*)   (smem + LDS_P1B);
    float*    bsum   = (float*)   (smem + LDS_BSUM);
    _Float16* chosen = (_Float16*)(smem + LDS_CHOSEN);
    _Float16* p1wl   = (_Float16*)(smem + LDS_P1W);
    _Float16* hid    = (_Float16*)(smem + LDS_UNION);   // aliases gates
    float*    gates  = (float*)   (smem + LDS_UNION);   // aliases hid

    int tid   = threadIdx.x;
    int wave  = tid >> 5;
    int lane  = tid & 31;
    int m16   = lane & 15;
    int khalf = lane >> 4;

    // ---- One-time init: state + step-invariant LDS caches ----
    for (int i = tid; i < BB * CC; i += 512) { h16[i] = (_Float16)0.f; cmem[i] = 0.f; }
    if (tid < BB)  accv[tid] = 0.f;
    if (tid < 128) { p2wl[tid] = p2w[tid]; p1bl[tid] = p1b[tid]; }
    bsum[tid] = bih[tid] + bhh[tid];                         // tid < 512 exactly
    for (int i = tid; i < (128 * 256) / 8; i += 512)         // p1_w -> LDS (v8h chunks)
        ((v8h*)p1wl)[i] = ((const v8h*)p1w16)[i];
    __syncthreads();

    for (int t = 1; t < LL; ++t) {
        const _Float16* cft = cf16 + (size_t)t * BB * FF;

        // Prefetch next step's cf slab (96KB) while this step computes.
        if (t + 1 < LL) {
            const char* nxt = (const char*)(cf16 + (size_t)(t + 1) * BB * FF);
            __builtin_prefetch(nxt + tid * 192, 0, 0);
        }

        // ---- Phase A: hid = relu([cf | h] @ p1_w^T + p1_b)  (384x128, K=256)
        for (int i = 0; i < 12; ++i) {
            int tile = wave * 12 + i;
            int rt = tile >> 3, ct = tile & 7;
            int r0 = rt * 16, c0 = ct * 16;
            int r  = r0 + m16;
            int b  = r / 6, kc = r - b * 6;
            const _Float16* arow_cf = cft  + (size_t)b * FF + kc * HH;
            const _Float16* arow_h  = h16  + b * CC;
            const _Float16* brow    = p1wl + (size_t)(c0 + m16) * 256;
            v8f acc = {};
#pragma unroll
            for (int kb = 0; kb < 128; kb += 32)            // K-half from cf (global)
                acc = WMMA(acc, frag_ld16(arow_cf + kb, khalf),
                                frag_ld16(brow + kb, khalf));
#pragma unroll
            for (int kb = 128; kb < 256; kb += 32)          // K-half from h (LDS)
                acc = WMMA(acc, frag_ld16(arow_h + (kb - 128), khalf),
                                frag_ld16(brow + kb, khalf));
#pragma unroll
            for (int v = 0; v < 8; ++v) {
                int mm = v + khalf * 8;
                float val = acc[v] + p1bl[c0 + m16];
                val = val > 0.f ? val : 0.f;
                hid[(size_t)(r0 + mm) * HH + c0 + m16] = (_Float16)val;
            }
        }
        __syncthreads();

        // ---- Phase B1: scores[b][k] = relu(hid . p2_w + p2_b), mask -> -inf
        if (tid < 384) {
            int b = tid / 6, kc = tid - b * 6;
            const _Float16* hr = hid + (size_t)tid * HH;
            float s = 0.f;
#pragma unroll
            for (int d0 = 0; d0 < HH; d0 += 8) {
                v8h hv = *(const v8h*)(hr + d0);
#pragma unroll
                for (int j = 0; j < 8; ++j) s += (float)hv[j] * p2wl[d0 + j];
            }
            s += p2b[0];
            s = s > 0.f ? s : 0.f;
            bool valid = maskp[(b * LL + t) * 6 + kc] != 0;
            scores[tid] = valid ? s : -1e30f;
        }
        __syncthreads();

        // ---- Phase B2: log-softmax + Gumbel-max sample
        if (tid < BB) {
            int b = tid;
            const float* sc = scores + b * 6;
            const float* gu = gumbel + ((size_t)(t - 1) * BB + b) * 6;
            float mx = -1e30f;
            for (int k = 0; k < 6; ++k) mx = sc[k] > mx ? sc[k] : mx;
            float se = 0.f;
            for (int k = 0; k < 6; ++k) se += expf(sc[k] - mx);
            float lse = mx + logf(se);
            int best = 0; float bv = -1e38f;
            for (int k = 0; k < 6; ++k) {
                float v = sc[k] + gu[k];
                if (v > bv) { bv = v; best = k; }
            }
            cidx[b] = best;
            out[b * TT + (t - 1)] = (float)best;
            if (lengthp[b] > t) accv[b] += sc[best] - lse;
        }
        __syncthreads();

        // ---- Phase C: chosen[b] = [cf(b,idx,0:128) | h(b) | h(b)]  (K=384 row)
        for (int i = tid; i < BB * 384; i += 512) {
            int b = i / 384, j = i - b * 384;
            _Float16 v;
            if (j < 128) v = cft[(size_t)b * FF + cidx[b] * HH + j];
            else         v = h16[b * CC + ((j - 128) & 127)];
            chosen[i] = v;
        }
        __syncthreads();

        // ---- Phase D: gates = chosen @ [w_ih | w_hh]^T + (b_ih + b_hh)
        for (int i = 0; i < 8; ++i) {
            int tile = wave * 8 + i;
            int rt = tile >> 5, ct = tile & 31;
            int r0 = rt * 16, c0 = ct * 16;
            const _Float16* arow  = chosen + (size_t)(r0 + m16) * 384;
            const _Float16* browi = wih16  + (size_t)(c0 + m16) * 256;
            const _Float16* browh = whh16  + (size_t)(c0 + m16) * 128;
            v8f acc = {};
#pragma unroll
            for (int kb = 0; kb < 256; kb += 32)            // K from w_ih
                acc = WMMA(acc, frag_ld16(arow + kb, khalf),
                                frag_ld16(browi + kb, khalf));
#pragma unroll
            for (int kb = 256; kb < 384; kb += 32)          // K from w_hh
                acc = WMMA(acc, frag_ld16(arow + kb, khalf),
                                frag_ld16(browh + (kb - 256), khalf));
#pragma unroll
            for (int v = 0; v < 8; ++v) {
                int mm = v + khalf * 8;
                gates[(size_t)(r0 + mm) * 512 + c0 + m16] = acc[v] + bsum[c0 + m16];
            }
        }
        __syncthreads();

        // ---- Phase E: LSTM cell
        for (int i = tid; i < BB * CC; i += 512) {
            int b = i >> 7, d = i & 127;
            float ig = gates[b * 512 + d];
            float fg = gates[b * 512 + 128 + d];
            float gg = gates[b * 512 + 256 + d];
            float og = gates[b * 512 + 384 + d];
            float si = 1.f / (1.f + expf(-ig));
            float sf = 1.f / (1.f + expf(-fg));
            float so = 1.f / (1.f + expf(-og));
            float cn = sf * cmem[i] + si * tanhf(gg);
            float hn = so * tanhf(cn);
            cmem[i] = cn;
            h16[i]  = (_Float16)hn;
        }
        __syncthreads();
    }

    if (tid < BB) out[BB * TT + tid] = accv[tid];
}

extern "C" void kernel_launch(void* const* d_in, const int* in_sizes, int n_in,
                              void* d_out, int out_size, void* d_ws, size_t ws_size,
                              hipStream_t stream) {
    const float* memory = (const float*)d_in[0];
    const int*   maskp  = (const int*)  d_in[1];
    const int*   length = (const int*)  d_in[2];
    const float* gumbel = (const float*)d_in[3];
    const float* l1_w   = (const float*)d_in[4];
    const float* l1_b   = (const float*)d_in[5];
    const float* p1_w   = (const float*)d_in[6];
    const float* p1_b   = (const float*)d_in[7];
    const float* p2_w   = (const float*)d_in[8];
    const float* p2_b   = (const float*)d_in[9];
    const float* w_ih   = (const float*)d_in[10];
    const float* w_hh   = (const float*)d_in[11];
    const float* b_ih   = (const float*)d_in[12];
    const float* b_hh   = (const float*)d_in[13];

    char* ws = (char*)d_ws;
    _Float16* cf16  = (_Float16*)(ws + WS_CF);
    _Float16* l1w16 = (_Float16*)(ws + WS_L1W);
    _Float16* p1w16 = (_Float16*)(ws + WS_P1W);
    _Float16* wih16 = (_Float16*)(ws + WS_WIH);
    _Float16* whh16 = (_Float16*)(ws + WS_WHH);

    cvt_f16_kernel<<<(FF * EE + 255) / 256, 256, 0, stream>>>(l1_w, l1w16, FF * EE);
    cvt_f16_kernel<<<(128 * 256 + 255) / 256, 256, 0, stream>>>(p1_w, p1w16, 128 * 256);
    cvt_f16_kernel<<<(512 * 256 + 255) / 256, 256, 0, stream>>>(w_ih, wih16, 512 * 256);
    cvt_f16_kernel<<<(512 * 128 + 255) / 256, 256, 0, stream>>>(w_hh, whh16, 512 * 128);

    cf_gemm_kernel<<<12288, 256, 0, stream>>>(memory, l1w16, l1_b, cf16);

    scan_kernel<<<1, 512, LDS_BYTES, stream>>>(cf16, p1w16, wih16, whh16,
                                               p1_b, p2_w, p2_b, b_ih, b_hh,
                                               maskp, length, gumbel,
                                               (float*)d_out);
}
